// CausalSelfAttention_12395275616281
// MI455X (gfx1250) — matrix operand order repaired and checked by reading, not compiled, not run on hardware
//
#include <hip/hip_runtime.h>

// ---------- CDNA5 (gfx1250, wave32) WMMA types ----------
typedef __attribute__((ext_vector_type(16))) _Float16 v16h;
typedef __attribute__((ext_vector_type(8)))  _Float16 v8h;
typedef __attribute__((ext_vector_type(4)))  _Float16 v4h;
typedef __attribute__((ext_vector_type(8)))  float    v8f;
typedef __attribute__((ext_vector_type(4)))  float    v4f;
typedef __attribute__((ext_vector_type(4)))  int      v4i;

union AFrag { v16h f; v8h h[2]; };

#define B_   4
#define T_   2048
#define C_   1024
#define H_   16
#define D_   64
#define MTOT (B_ * T_)   // 8192

// ---- gfx1250 async global->LDS path (ASYNCcnt), with safe fallback --------
#if defined(__AMDGCN__) && __has_builtin(__builtin_amdgcn_global_load_async_to_lds_b128)
#define ASYNC_LDS 1
#else
#define ASYNC_LDS 0
#endif

__device__ __forceinline__ void cp16_async(_Float16* lds_dst, const _Float16* gsrc) {
#if ASYNC_LDS
  __builtin_amdgcn_global_load_async_to_lds_b128(
      (__attribute__((address_space(1))) v4i*)gsrc,
      (__attribute__((address_space(3))) v4i*)lds_dst,
      /*imm offset*/0, /*cpol*/0);
#else
  *(v8h*)lds_dst = *(const v8h*)gsrc;   // VGPR bounce fallback
#endif
}

__device__ __forceinline__ void async_wait0() {
#if ASYNC_LDS
#if __has_builtin(__builtin_amdgcn_s_wait_asynccnt)
  __builtin_amdgcn_s_wait_asynccnt(0);
#else
  asm volatile("s_wait_asynccnt 0x0" ::: "memory");
#endif
#endif
}

__device__ __forceinline__ v8f vzero8() {
  v8f z = {0.f, 0.f, 0.f, 0.f, 0.f, 0.f, 0.f, 0.f};
  return z;
}

__device__ __forceinline__ v8f wmma16(const AFrag& a, const AFrag& b, v8f c) {
  // D = A(16x32 f16) * B(32x16 f16) + C(16x16 f32)
  return __builtin_amdgcn_wmma_f32_16x16x32_f16(
      /*neg_a=*/false, a.f, /*neg_b=*/false, b.f,
      /*c_mod=*/(short)0, c, /*reuse_a=*/false, /*reuse_b=*/false);
}

// ---------------------------------------------------------------------------
// f32 -> f16 convert (vector x4)
// ---------------------------------------------------------------------------
__global__ void cvt4_kernel(const float* __restrict__ in,
                            _Float16* __restrict__ out, int n4) {
  int i = blockIdx.x * blockDim.x + threadIdx.x;
  if (i < n4) {
    v4f a = ((const v4f*)in)[i];
    v4h o;
    o.x = (_Float16)a.x; o.y = (_Float16)a.y;
    o.z = (_Float16)a.z; o.w = (_Float16)a.w;
    ((v4h*)out)[i] = o;
  }
}

// ---------------------------------------------------------------------------
// W[K][N] f32  ->  Wt[N][K] f16  (LDS tile transpose)
// ---------------------------------------------------------------------------
__global__ void transpose_f16_kernel(const float* __restrict__ W,
                                     _Float16* __restrict__ Wt, int K, int N) {
  __shared__ float tile[32][33];
  const int kb = blockIdx.y * 32;
  const int nb = blockIdx.x * 32;
  const int tx = threadIdx.x;   // 32
  const int ty = threadIdx.y;   // 8
  #pragma unroll
  for (int i = ty; i < 32; i += 8)
    tile[i][tx] = W[(size_t)(kb + i) * N + nb + tx];
  __syncthreads();
  #pragma unroll
  for (int i = ty; i < 32; i += 8)
    Wt[(size_t)(nb + i) * K + kb + tx] = (_Float16)tile[tx][i];
}

// ---------------------------------------------------------------------------
// WMMA GEMM: Y = A[M][K]f16 * Bt[N][K]f16^T + bias
//   mode 0: scatter epilogue into q[B,H,T,D] / k[B,H,T,D] / v[B,H,D,T] (f16)
//   mode 1: dense f32 output [M][N]
// Block: 256 thr (8 waves). Block tile 128x128, wave tile 32x64, BK=32.
// LDS double-buffered; tiles fetched with async global->LDS DMA (ASYNCcnt).
// ---------------------------------------------------------------------------
#define BM  128
#define BN  128
#define BK  32
#define LDT 40   // BK + 8 halfs -> 80B row stride (16B aligned, conflict-spread)

__launch_bounds__(256)
__global__ void gemm_kernel(const _Float16* __restrict__ A,
                            const _Float16* __restrict__ Bt,
                            const float* __restrict__ bias,
                            int M, int N, int K, int mode,
                            _Float16* __restrict__ qo,
                            _Float16* __restrict__ ko,
                            _Float16* __restrict__ vo,
                            float* __restrict__ fo) {
  __shared__ _Float16 As[2][BM * LDT];
  __shared__ _Float16 Bs[2][BN * LDT];

  const int tid  = threadIdx.x;
  const int lane = tid & 31;
  const int w    = tid >> 5;          // wave 0..7
  const int hf   = lane >> 4;         // half-wave 0/1
  const int lr   = lane & 15;
  const int wm   = (w >> 1) * 32;     // wave M offset in block tile
  const int wn   = (w & 1)  * 64;     // wave N offset in block tile

  const int m0 = blockIdx.y * BM;
  const int n0 = blockIdx.x * BN;

  // global->LDS: 256 threads x 16 halfs per matrix tile
  const int lrow = tid >> 1;          // 0..127
  const int lcol = (tid & 1) * 16;    // 0 / 16
  const int lofs = lrow * LDT + lcol;

  const _Float16* Agb = A  + (size_t)(m0 + lrow) * K + lcol;
  const _Float16* Bgb = Bt + (size_t)(n0 + lrow) * K + lcol;

  // prologue: issue tile 0 into buffer 0
  cp16_async(&As[0][lofs],     Agb);
  cp16_async(&As[0][lofs + 8], Agb + 8);
  cp16_async(&Bs[0][lofs],     Bgb);
  cp16_async(&Bs[0][lofs + 8], Bgb + 8);
  async_wait0();
  __syncthreads();

  v8f acc[2][4];
  #pragma unroll
  for (int mi = 0; mi < 2; ++mi)
    #pragma unroll
    for (int ni = 0; ni < 4; ++ni) acc[mi][ni] = vzero8();

  int cur = 0;
  for (int k0 = 0; k0 < K; k0 += BK) {
    const int nxt = cur ^ 1;
    // issue next tile while this tile computes
    if (k0 + BK < K) {
      const _Float16* An = Agb + k0 + BK;
      const _Float16* Bn = Bgb + k0 + BK;
      cp16_async(&As[nxt][lofs],     An);
      cp16_async(&As[nxt][lofs + 8], An + 8);
      cp16_async(&Bs[nxt][lofs],     Bn);
      cp16_async(&Bs[nxt][lofs + 8], Bn + 8);
      __builtin_prefetch(An + BK, 0, 3);   // gfx1250 global_prefetch_b8
      __builtin_prefetch(Bn + BK, 0, 3);
    }

    AFrag af[2], bf[4];
    #pragma unroll
    for (int mi = 0; mi < 2; ++mi) {
      // ISA 16-bit A layout: lane(hf,lr) holds row lr, K chunks hf*8 and 16+hf*8
      const _Float16* p = &As[cur][(wm + mi * 16 + lr) * LDT];
      af[mi].h[0] = *(const v8h*)(p + hf * 8);
      af[mi].h[1] = *(const v8h*)(p + 16 + hf * 8);
    }
    #pragma unroll
    for (int ni = 0; ni < 4; ++ni) {
      // ISA 16-bit B layout: lane(hf,lr) holds col lr, K chunks hf*8 and 16+hf*8
      const _Float16* p = &Bs[cur][(wn + ni * 16 + lr) * LDT];
      bf[ni].h[0] = *(const v8h*)(p + hf * 8);
      bf[ni].h[1] = *(const v8h*)(p + 16 + hf * 8);
    }
    #pragma unroll
    for (int mi = 0; mi < 2; ++mi)
      #pragma unroll
      for (int ni = 0; ni < 4; ++ni)
        acc[mi][ni] = wmma16(af[mi], bf[ni], acc[mi][ni]);

    async_wait0();      // next tile landed in LDS
    __syncthreads();    // publish across waves
    cur = nxt;
  }

  // epilogue: C/D layout -> element (row = 8*hf + j, col = lr) per 16x16 tile
  #pragma unroll
  for (int mi = 0; mi < 2; ++mi) {
    #pragma unroll
    for (int ni = 0; ni < 4; ++ni) {
      const int gn = n0 + wn + ni * 16 + lr;
      const float bv = bias[gn];
      #pragma unroll
      for (int j = 0; j < 8; ++j) {
        const int gm = m0 + wm + mi * 16 + hf * 8 + j;
        const float val = acc[mi][ni][j] + bv;
        if (mode == 1) {
          fo[(size_t)gm * N + gn] = val;
        } else {
          const _Float16 hv = (_Float16)val;
          const int which = gn >> 10;        // 0:q 1:k 2:v
          const int cc = gn & 1023;
          const int hd = cc >> 6, dd = cc & 63;
          const int bi = gm >> 11, tt = gm & 2047;
          if (which == 0)
            qo[(((size_t)bi * H_ + hd) * T_ + tt) * D_ + dd] = hv;
          else if (which == 1)
            ko[(((size_t)bi * H_ + hd) * T_ + tt) * D_ + dd] = hv;
          else  // V stored transposed [B,H,D,T] for PV B-fragments
            vo[(((size_t)bi * H_ + hd) * D_ + dd) * T_ + tt] = hv;
        }
      }
    }
  }
}

// ---------------------------------------------------------------------------
// Flash attention, causal. Block = (b, h, 128 query rows), 8 waves x 16 rows.
// Key blocks of 64; K/V tiles double-buffered in LDS via async DMA;
// P bounced through per-wave LDS scratch to reach A-fragment layout.
// ---------------------------------------------------------------------------
#define KB  64
#define LKD 72   // Ks row stride (64 d + pad 8) -> 144B
#define LVK 72   // Vs row stride (64 k + pad 8)
#define LPK 80   // Ps row stride (64 k + pad 16) -> 160B

__launch_bounds__(256)
__global__ void attn_kernel(const _Float16* __restrict__ q,   // [B,H,T,D]
                            const _Float16* __restrict__ k,   // [B,H,T,D]
                            const _Float16* __restrict__ v,   // [B,H,D,T]
                            _Float16* __restrict__ y) {       // [B*T, C]
  __shared__ _Float16 Ks[2][KB * LKD];
  __shared__ _Float16 Vs[2][D_ * LVK];
  __shared__ _Float16 Ps[8 * 16 * LPK];   // per-wave 16 x 64 P scratch

  const int qb = blockIdx.x;   // query block (128 rows)
  const int hh = blockIdx.y;
  const int bb = blockIdx.z;

  const int tid  = threadIdx.x;
  const int lane = tid & 31;
  const int w    = tid >> 5;
  const int hf   = lane >> 4;
  const int lr   = lane & 15;

  const size_t bh = (size_t)bb * H_ + hh;
  const _Float16* qp = q + bh * T_ * D_;
  const _Float16* kp = k + bh * T_ * D_;
  const _Float16* vp = v + bh * D_ * T_;

  const int qrow0 = qb * 128 + w * 16;     // this wave's 16 query rows

  // Q fragments (d-dim 64 => 2 A-frags), loaded once from global
  AFrag qa[2];
  {
    const _Float16* qr = qp + (size_t)(qrow0 + lr) * D_;
    #pragma unroll
    for (int kk = 0; kk < 2; ++kk) {
      qa[kk].h[0] = *(const v8h*)(qr + kk * 32 + hf * 8);
      qa[kk].h[1] = *(const v8h*)(qr + kk * 32 + 16 + hf * 8);
    }
  }

  v8f o[4];
  #pragma unroll
  for (int dt = 0; dt < 4; ++dt) o[dt] = vzero8();
  float mrow[8], lsum[8];
  #pragma unroll
  for (int j = 0; j < 8; ++j) { mrow[j] = -__builtin_inff(); lsum[j] = 0.f; }

  const float scale = 0.125f;   // 1/sqrt(64)

  // K/V tile load map: 64 rows, 4 threads/row, 16 halfs each
  const int trow = tid >> 2;
  const int tcol = (tid & 3) * 16;

  // issue K/V tile for key block at kbase into buffer buf
  auto issue_kv = [&](int buf, int kbase) {
    const _Float16* kg = kp + (size_t)(kbase + trow) * D_ + tcol;
    cp16_async(&Ks[buf][trow * LKD + tcol],     kg);
    cp16_async(&Ks[buf][trow * LKD + tcol + 8], kg + 8);
    const _Float16* vg = vp + (size_t)trow * T_ + kbase + tcol;
    cp16_async(&Vs[buf][trow * LVK + tcol],     vg);
    cp16_async(&Vs[buf][trow * LVK + tcol + 8], vg + 8);
  };

  const int kbmax = 2 * qb + 1;
  issue_kv(0, 0);
  async_wait0();
  __syncthreads();

  int cur = 0;
  for (int kbi = 0; kbi <= kbmax; ++kbi) {
    const int kbase = kbi * KB;
    const int nxt = cur ^ 1;
    if (kbi < kbmax) issue_kv(nxt, kbase + KB);   // overlap next K/V tile

    if (kbase <= qrow0 + 15) {   // wave-uniform: skip fully-masked key blocks
      // ---- S = Q K^T (4 key tiles x 2 k-chunks) ----
      v8f st[4];
      #pragma unroll
      for (int n = 0; n < 4; ++n) st[n] = vzero8();
      #pragma unroll
      for (int kk = 0; kk < 2; ++kk) {
        #pragma unroll
        for (int n = 0; n < 4; ++n) {
          AFrag bf;
          const _Float16* p = &Ks[cur][(n * 16 + lr) * LKD + kk * 32];
          bf.h[0] = *(const v8h*)(p + hf * 8);
          bf.h[1] = *(const v8h*)(p + 16 + hf * 8);
          st[n] = wmma16(qa[kk], bf, st[n]);
        }
      }

      // ---- scale + causal mask ----
      float sv[4][8];
      #pragma unroll
      for (int n = 0; n < 4; ++n) {
        const int kpos = kbase + n * 16 + lr;
        #pragma unroll
        for (int j = 0; j < 8; ++j) {
          const int qpos = qrow0 + hf * 8 + j;
          float x = st[n][j] * scale;
          sv[n][j] = (kpos > qpos) ? -__builtin_inff() : x;
        }
      }

      // ---- online softmax (rows live in 16-lane halves; xor-shuffle reduce)
      float corr[8];
      #pragma unroll
      for (int j = 0; j < 8; ++j) {
        float rm = fmaxf(fmaxf(sv[0][j], sv[1][j]), fmaxf(sv[2][j], sv[3][j]));
        #pragma unroll
        for (int s = 1; s < 16; s <<= 1) rm = fmaxf(rm, __shfl_xor(rm, s, 32));
        const float mnew = fmaxf(mrow[j], rm);
        corr[j] = __expf(mrow[j] - mnew);
        float rs = 0.f;
        #pragma unroll
        for (int n = 0; n < 4; ++n) {
          const float p = __expf(sv[n][j] - mnew);
          sv[n][j] = p;
          rs += p;
        }
        #pragma unroll
        for (int s = 1; s < 16; s <<= 1) rs += __shfl_xor(rs, s, 32);
        lsum[j] = lsum[j] * corr[j] + rs;
        mrow[j] = mnew;
        #pragma unroll
        for (int dt = 0; dt < 4; ++dt) o[dt][j] *= corr[j];
      }

      // ---- P (C-layout) -> per-wave LDS -> A-fragments ----
      _Float16* pw = &Ps[w * 16 * LPK];
      #pragma unroll
      for (int n = 0; n < 4; ++n)
        #pragma unroll
        for (int j = 0; j < 8; ++j)
          pw[(hf * 8 + j) * LPK + n * 16 + lr] = (_Float16)sv[n][j];
      // same-wave LDS ops are in-order: no barrier needed (private scratch)

      // ---- O += P V ----
      #pragma unroll
      for (int kk = 0; kk < 2; ++kk) {
        AFrag pa;
        const _Float16* pr = &pw[lr * LPK + kk * 32];
        pa.h[0] = *(const v8h*)(pr + hf * 8);
        pa.h[1] = *(const v8h*)(pr + 16 + hf * 8);
        #pragma unroll
        for (int dt = 0; dt < 4; ++dt) {
          AFrag vb;
          const _Float16* vr = &Vs[cur][(dt * 16 + lr) * LVK + kk * 32];
          vb.h[0] = *(const v8h*)(vr + hf * 8);
          vb.h[1] = *(const v8h*)(vr + 16 + hf * 8);
          o[dt] = wmma16(pa, vb, o[dt]);
        }
      }
    }

    async_wait0();      // next K/V tile landed
    __syncthreads();
    cur = nxt;
  }

  // ---- normalize + write y[b*T + t][h*64 + d] ----
  #pragma unroll
  for (int j = 0; j < 8; ++j) {
    const float inv = 1.0f / lsum[j];
    const int tt = qrow0 + hf * 8 + j;
    const size_t rowoff = ((size_t)bb * T_ + tt) * C_ + hh * D_;
    #pragma unroll
    for (int dt = 0; dt < 4; ++dt)
      y[rowoff + dt * 16 + lr] = (_Float16)(o[dt][j] * inv);
  }
}

// ---------------------------------------------------------------------------
extern "C" void kernel_launch(void* const* d_in, const int* in_sizes, int n_in,
                              void* d_out, int out_size, void* d_ws, size_t ws_size,
                              hipStream_t stream) {
  const float* x     = (const float*)d_in[0];
  const float* Wqkv  = (const float*)d_in[1];
  const float* bqkv  = (const float*)d_in[2];
  const float* Wproj = (const float*)d_in[3];
  const float* bproj = (const float*)d_in[4];
  float* out = (float*)d_out;

  _Float16* ws = (_Float16*)d_ws;
  size_t off = 0;
  _Float16* xh     = ws + off; off += (size_t)MTOT * C_;        // 8M halfs
  _Float16* wqkvt  = ws + off; off += (size_t)(3 * C_) * C_;    // 3M
  _Float16* wprojt = ws + off; off += (size_t)C_ * C_;          // 1M
  _Float16* qbuf   = ws + off; off += (size_t)B_ * H_ * T_ * D_;
  _Float16* kbuf   = ws + off; off += (size_t)B_ * H_ * T_ * D_;
  _Float16* vbuf   = ws + off; off += (size_t)B_ * H_ * T_ * D_;
  _Float16* yh     = ws + off; off += (size_t)MTOT * C_;
  (void)ws_size; (void)in_sizes; (void)n_in; (void)out_size;

  // 1) convert x to f16
  {
    const int n4 = (MTOT * C_) / 4;
    cvt4_kernel<<<(n4 + 255) / 256, 256, 0, stream>>>(x, xh, n4);
  }
  // 2) transpose+convert weights to [N][K] f16
  transpose_f16_kernel<<<dim3((3 * C_) / 32, C_ / 32), dim3(32, 8), 0, stream>>>(
      Wqkv, wqkvt, C_, 3 * C_);
  transpose_f16_kernel<<<dim3(C_ / 32, C_ / 32), dim3(32, 8), 0, stream>>>(
      Wproj, wprojt, C_, C_);
  // 3) QKV projection (scatter epilogue into q/k/v)
  gemm_kernel<<<dim3((3 * C_) / BN, MTOT / BM), 256, 0, stream>>>(
      xh, wqkvt, bqkv, MTOT, 3 * C_, C_, /*mode=*/0, qbuf, kbuf, vbuf, nullptr);
  // 4) causal flash attention
  attn_kernel<<<dim3(T_ / 128, H_, B_), 256, 0, stream>>>(qbuf, kbuf, vbuf, yh);
  // 5) output projection (f32 epilogue)
  gemm_kernel<<<dim3(C_ / BN, MTOT / BM), 256, 0, stream>>>(
      yh, wprojt, bproj, MTOT, C_, C_, /*mode=*/1, nullptr, nullptr, nullptr, out);
}
